// MultiHeadAttention_2645699854443
// MI455X (gfx1250) — matrix operand order repaired and checked by reading, not compiled, not run on hardware
//
#include <hip/hip_runtime.h>
#include <hip/hip_bf16.h>

// ---------------------------------------------------------------------------
// MHA forward for B=2, S=2048, D=768, H=12, DK=64 on gfx1250 (wave32, WMMA).
// bf16 WMMA (f32 accumulate) everywhere; flash-attention single pass (no
// 2*12*2048^2 score tensor in HBM). K-chunks of 64 to amortize barriers and
// softmax VALU over 16 WMMAs per LDS stage. bf16-sourced tiles use the CDNA5
// async global->LDS path (ASYNCcnt) when the builtins are available.
// ---------------------------------------------------------------------------

typedef __bf16 bf16;
typedef bf16  v16bf __attribute__((ext_vector_type(16)));
typedef bf16  v8bf  __attribute__((ext_vector_type(8)));
typedef bf16  v4bf  __attribute__((ext_vector_type(4)));
typedef float v8f   __attribute__((ext_vector_type(8)));
typedef int   i32x4 __attribute__((ext_vector_type(4)));

// AS-qualified pointee types for the typed async-copy builtin
typedef __attribute__((address_space(1))) i32x4 g_i32x4;   // global
typedef __attribute__((address_space(3))) i32x4 l_i32x4;   // LDS

#define WMMA_BF16(a, b, c) \
  __builtin_amdgcn_wmma_f32_16x16x32_bf16(false, (a), false, (b), (short)0, (c), false, false)

#if defined(__gfx1250__) && __has_builtin(__builtin_amdgcn_global_load_async_to_lds_b128)
#define ASYNC_LDS 1
#else
#define ASYNC_LDS 0
#endif

__device__ __forceinline__ void async_ld16(void* lds, const void* g) {
#if ASYNC_LDS
  __builtin_amdgcn_global_load_async_to_lds_b128(
      (g_i32x4*)(unsigned long long)(uintptr_t)g,
      (l_i32x4*)(unsigned)(uintptr_t)lds, 0, 0);
#else
  (void)lds; (void)g;
#endif
}

__device__ __forceinline__ void wait_async0() {
#if ASYNC_LDS
#if __has_builtin(__builtin_amdgcn_s_wait_asynccnt)
  __builtin_amdgcn_s_wait_asynccnt(0);
#else
  asm volatile("s_wait_asynccnt 0x0" ::: "memory");
#endif
#endif
}

constexpr int kB  = 2;
constexpr int kS  = 2048;
constexpr int kD  = 768;
constexpr int kH  = 12;
constexpr int kDK = 64;
constexpr int kM  = kB * kS;                    // 4096
constexpr int kHeadElems = kB * kH * kS * kDK;  // bf16 elems per head tensor

__device__ __forceinline__ float toF(float x) { return x; }
__device__ __forceinline__ float toF(bf16 x)  { return (float)x; }

// A-matrix fragment (16x32 bf16), ISA 7.12.2 layout.
__device__ __forceinline__ v16bf load_frag_A(const bf16* base, int ld, int lane) {
  const int row = lane & 15;
  const int kh  = lane >> 4;
  const v8bf lo = *(const v8bf*)(base + row * ld + kh * 8);
  const v8bf hi = *(const v8bf*)(base + row * ld + 16 + kh * 8);
  v16bf a;
#pragma unroll
  for (int j = 0; j < 8; ++j) { a[j] = lo[j]; a[8 + j] = hi[j]; }
  return a;
}

// B-matrix fragment (32x16 bf16), tile stored [n][k] with k contiguous.
__device__ __forceinline__ v16bf load_frag_B(const bf16* baseNk, int ld, int lane) {
  const int n  = lane & 15;
  const int kh = lane >> 4;
  const bf16* p = baseNk + n * ld + kh * 8;
  const v8bf lo = *(const v8bf*)(p);
  const v8bf hi = *(const v8bf*)(p + 16);
  v16bf b;
#pragma unroll
  for (int j = 0; j < 8; ++j) { b[j] = lo[j]; b[8 + j] = hi[j]; }
  return b;
}

// ---------------------------------------------------------------------------
// Projection GEMM: Y[4096 x 768] = X @ W^T + bias (W is [out,in]).
// Block = 256 threads (8 waves), tile 64(M) x 128(N), K-chunks of 64.
// OUT_MODE 0: bf16 -> [B,H,S,DK] head layout.  OUT_MODE 1: f32 row-major.
// ---------------------------------------------------------------------------
template <int OUT_MODE, typename XT>
__global__ __launch_bounds__(256) void gemm_proj(const XT* __restrict__ X,
                                                 const float* __restrict__ W,
                                                 const float* __restrict__ bias,
                                                 void* __restrict__ Yout) {
  constexpr int K   = kD;
  constexpr int CH  = 64;   // K chunk
  constexpr int LDX = 72;   // 64 + 8 bf16 pad (144B row stride, 16B aligned)
  constexpr bool kAsyncX = ASYNC_LDS && (sizeof(XT) == 2);

  __shared__ __attribute__((aligned(16))) bf16 Xs[64 * LDX];
  __shared__ __attribute__((aligned(16))) bf16 Ws[128 * LDX];

  const int tid  = threadIdx.x;
  const int lane = tid & 31;
  const int wid  = tid >> 5;
  const int wm   = (wid >> 2) * 32;
  const int wn   = (wid & 3) * 32;
  const int n0   = blockIdx.x * 128;
  const int m0   = blockIdx.y * 64;

  v8f acc[2][2];
#pragma unroll
  for (int i = 0; i < 2; ++i)
#pragma unroll
    for (int j = 0; j < 2; ++j) acc[i][j] = {};

  for (int kk = 0; kk < K; kk += CH) {
    // ---- X tile: 64 rows x 64 k ----
    if constexpr (kAsyncX) {
      // bf16 source: async copy straight into LDS (512 x 16B chunks).
#pragma unroll
      for (int i = 0; i < 2; ++i) {
        const int cid = i * 256 + tid;
        const int r = cid >> 3, c8 = (cid & 7) * 8;
        async_ld16(Xs + r * LDX + c8,
                   (const bf16*)X + (size_t)(m0 + r) * K + kk + c8);
      }
    } else {
      // fp32 source: load + convert (4096 elems, 4 iters x v4 per thread).
#pragma unroll
      for (int i = 0; i < 4; ++i) {
        const int e = i * 256 + tid;
        const int r = e >> 4, c4 = (e & 15) * 4;
        const XT* src = X + (size_t)(m0 + r) * K + kk + c4;
        v4bf o;
#pragma unroll
        for (int j = 0; j < 4; ++j) o[j] = (bf16)toF(src[j]);
        *(v4bf*)(Xs + r * LDX + c4) = o;
      }
    }
    // ---- W tile as B-layout [n][k]: 128 x 64 (8192 elems, convert) ----
#pragma unroll
    for (int i = 0; i < 8; ++i) {
      const int e = i * 256 + tid;
      const int n = e >> 4, c4 = (e & 15) * 4;
      const float* src = W + (size_t)(n0 + n) * K + kk + c4;
      v4bf o;
#pragma unroll
      for (int j = 0; j < 4; ++j) o[j] = (bf16)src[j];
      *(v4bf*)(Ws + n * LDX + c4) = o;
    }
    // prefetch next K chunk (speculative; dropped if OOB)
    {
      const int nkk = (kk + CH < K) ? kk + CH : kk;
      __builtin_prefetch(W + (size_t)(n0 + (tid >> 1)) * K + nkk + (tid & 1) * 32, 0, 0);
    }
    wait_async0();
    __syncthreads();

    v16bf a[2][2], b[2][2];
#pragma unroll
    for (int i = 0; i < 2; ++i)
#pragma unroll
      for (int ch = 0; ch < 2; ++ch)
        a[i][ch] = load_frag_A(Xs + (wm + i * 16) * LDX + ch * 32, LDX, lane);
#pragma unroll
    for (int j = 0; j < 2; ++j)
#pragma unroll
      for (int ch = 0; ch < 2; ++ch)
        b[j][ch] = load_frag_B(Ws + (wn + j * 16) * LDX + ch * 32, LDX, lane);
#pragma unroll
    for (int ch = 0; ch < 2; ++ch)
#pragma unroll
      for (int i = 0; i < 2; ++i)
#pragma unroll
        for (int j = 0; j < 2; ++j) acc[i][j] = WMMA_BF16(a[i][ch], b[j][ch], acc[i][j]);

    __syncthreads();
  }

  // Epilogue. C layout: N = lane&15, M = v + 8*(lane>>4).
#pragma unroll
  for (int i = 0; i < 2; ++i) {
#pragma unroll
    for (int j = 0; j < 2; ++j) {
      const int n    = n0 + wn + j * 16 + (lane & 15);
      const float bv = bias[n];
      const int mb   = m0 + wm + i * 16 + 8 * (lane >> 4);
#pragma unroll
      for (int v = 0; v < 8; ++v) {
        const int m   = mb + v;
        const float y = acc[i][j][v] + bv;
        if (OUT_MODE == 0) {
          const int b = m >> 11, s = m & (kS - 1);
          const int h = n >> 6, dk = n & 63;
          ((bf16*)Yout)[(((size_t)b * kH + h) * kS + s) * kDK + dk] = (bf16)y;
        } else {
          ((float*)Yout)[(size_t)m * kD + n] = y;
        }
      }
    }
  }
}

// ---------------------------------------------------------------------------
// Flash attention: block = (b*H+h, 64-query tile), 128 threads (4 waves),
// each wave owns 16 query rows. Key chunks of 64. Online softmax with row
// stats replicated across each 16-lane half group; P through per-wave LDS to
// convert C-layout -> A-layout for P@V.
// ---------------------------------------------------------------------------
__global__ __launch_bounds__(128) void attn_kernel(const bf16* __restrict__ Qh,
                                                   const bf16* __restrict__ Kh,
                                                   const bf16* __restrict__ Vh,
                                                   bf16* __restrict__ Xa) {
  constexpr int CH  = 64;   // keys per chunk
  constexpr int LDQ = 72;
  constexpr int LDK = 72;
  constexpr int LDV = 72;   // 64 keys + 8 pad
  constexpr int LDP = 72;

  __shared__ __attribute__((aligned(16))) bf16 Qs[64 * LDQ];
  __shared__ __attribute__((aligned(16))) bf16 Ks[64 * LDK];   // [key][dk]
  __shared__ __attribute__((aligned(16))) bf16 Vs[64 * LDV];   // [dk][key] (transposed)
  __shared__ __attribute__((aligned(16))) bf16 Ps[4 * 16 * LDP];

  const int tid  = threadIdx.x;
  const int lane = tid & 31;
  const int wid  = tid >> 5;
  const int bh   = blockIdx.y;          // b*H + h
  const int b    = bh / kH;
  const int h    = bh - b * kH;
  const int q0   = blockIdx.x * 64;

  // ---- 64x64 Q tile ----
  const size_t qbase = ((size_t)bh * kS + q0) * kDK;
#if ASYNC_LDS
#pragma unroll
  for (int i = 0; i < 4; ++i) {
    const int cid = i * 128 + tid;
    const int r = cid >> 3, c8 = (cid & 7) * 8;
    async_ld16(Qs + r * LDQ + c8, Qh + qbase + r * kDK + c8);
  }
  wait_async0();
#else
#pragma unroll
  for (int i = 0; i < 8; ++i) {
    const int e = i * 128 + tid;
    const int r = e >> 4, c4 = (e & 15) * 4;
    *(v4bf*)(Qs + r * LDQ + c4) = *(const v4bf*)(Qh + qbase + r * kDK + c4);
  }
#endif
  __syncthreads();

  v16bf aQ[2];
#pragma unroll
  for (int ch = 0; ch < 2; ++ch)
    aQ[ch] = load_frag_A(Qs + (wid * 16) * LDQ + ch * 32, LDQ, lane);

  float mrow[8], lrow[8];
  v8f o[4];
#pragma unroll
  for (int v = 0; v < 8; ++v) { mrow[v] = -__builtin_inff(); lrow[v] = 0.0f; }
#pragma unroll
  for (int nt = 0; nt < 4; ++nt) o[nt] = {};

  const float scale = 0.125f;  // 1/sqrt(64)
  bf16* Pw = Ps + wid * 16 * LDP;

  for (int kt = 0; kt < kS; kt += CH) {
    const size_t kb = ((size_t)bh * kS + kt) * kDK;
    // K tile 64x64 -> Ks[key][dk] (async when available)
#if ASYNC_LDS
#pragma unroll
    for (int i = 0; i < 4; ++i) {
      const int cid = i * 128 + tid;
      const int r = cid >> 3, c8 = (cid & 7) * 8;
      async_ld16(Ks + r * LDK + c8, Kh + kb + r * kDK + c8);
    }
#else
#pragma unroll
    for (int i = 0; i < 8; ++i) {
      const int e = i * 128 + tid;
      const int r = e >> 4, c4 = (e & 15) * 4;
      *(v4bf*)(Ks + r * LDK + c4) = *(const v4bf*)(Kh + kb + r * kDK + c4);
    }
#endif
    // V tile 64x64 -> Vs[dk][key] (transposed through VGPRs)
#pragma unroll
    for (int i = 0; i < 8; ++i) {
      const int e = i * 128 + tid;
      const int r = e >> 4, c4 = (e & 15) * 4;
      const v4bf vv = *(const v4bf*)(Vh + kb + r * kDK + c4);
#pragma unroll
      for (int j = 0; j < 4; ++j) Vs[(c4 + j) * LDV + r] = vv[j];
    }
    // prefetch next chunk
    if (kt + CH < kS)
      __builtin_prefetch(Kh + kb + (size_t)CH * kDK + tid * 32, 0, 0);
    wait_async0();
    __syncthreads();

    // scores: S[16 x 64] = Q(16x64) @ K^T(64x64)
    v8f sc[4];
#pragma unroll
    for (int nt = 0; nt < 4; ++nt) sc[nt] = {};
#pragma unroll
    for (int nt = 0; nt < 4; ++nt)
#pragma unroll
      for (int ch = 0; ch < 2; ++ch) {
        const v16bf bK = load_frag_B(Ks + (nt * 16) * LDK + ch * 32, LDK, lane);
        sc[nt] = WMMA_BF16(aQ[ch], bK, sc[nt]);
      }

    // online softmax per row (row = v + 8*(lane>>4); N striped over 16 lanes)
#pragma unroll
    for (int v = 0; v < 8; ++v) {
      float s[4];
#pragma unroll
      for (int nt = 0; nt < 4; ++nt) s[nt] = sc[nt][v] * scale;
      float rm = fmaxf(fmaxf(s[0], s[1]), fmaxf(s[2], s[3]));
#pragma unroll
      for (int off = 1; off < 16; off <<= 1) rm = fmaxf(rm, __shfl_xor(rm, off, 32));
      const float mnew = fmaxf(mrow[v], rm);
      const float corr = __expf(mrow[v] - mnew);
      float p[4], rs = 0.0f;
#pragma unroll
      for (int nt = 0; nt < 4; ++nt) { p[nt] = __expf(s[nt] - mnew); rs += p[nt]; }
#pragma unroll
      for (int off = 1; off < 16; off <<= 1) rs += __shfl_xor(rs, off, 32);
      lrow[v] = lrow[v] * corr + rs;
      mrow[v] = mnew;
#pragma unroll
      for (int nt = 0; nt < 4; ++nt) o[nt][v] *= corr;
      const int prow = v + 8 * (lane >> 4);
#pragma unroll
      for (int nt = 0; nt < 4; ++nt)
        Pw[prow * LDP + nt * 16 + (lane & 15)] = (bf16)p[nt];
    }

    // O(16x64) += P(16x64) @ V(64x64). LDS ops are in-order within a wave,
    // so the Pw stores above are visible to these loads without a barrier.
    v16bf aP[2];
#pragma unroll
    for (int kc = 0; kc < 2; ++kc) aP[kc] = load_frag_A(Pw + kc * 32, LDP, lane);
#pragma unroll
    for (int nt = 0; nt < 4; ++nt)
#pragma unroll
      for (int kc = 0; kc < 2; ++kc) {
        const v16bf bV = load_frag_B(Vs + (nt * 16) * LDV + kc * 32, LDV, lane);
        o[nt] = WMMA_BF16(aP[kc], bV, o[nt]);
      }
    __syncthreads();  // before next chunk overwrites Ks/Vs
  }

  // normalize and write merged-head layout [B, S, D] (bf16 for final GEMM)
#pragma unroll
  for (int v = 0; v < 8; ++v) {
    const float inv = 1.0f / lrow[v];
    const int row = wid * 16 + v + 8 * (lane >> 4);
    const int s = q0 + row;
#pragma unroll
    for (int nt = 0; nt < 4; ++nt) {
      const int col = h * kDK + nt * 16 + (lane & 15);
      Xa[((size_t)b * kS + s) * kD + col] = (bf16)(o[nt][v] * inv);
    }
  }
}

// ---------------------------------------------------------------------------
extern "C" void kernel_launch(void* const* d_in, const int* in_sizes, int n_in,
                              void* d_out, int out_size, void* d_ws, size_t ws_size,
                              hipStream_t stream) {
  const float* q  = (const float*)d_in[0];
  const float* k  = (const float*)d_in[1];
  const float* v  = (const float*)d_in[2];
  const float* Wq = (const float*)d_in[3];
  const float* bq = (const float*)d_in[4];
  const float* Wk = (const float*)d_in[5];
  const float* bk = (const float*)d_in[6];
  const float* Wv = (const float*)d_in[7];
  const float* bv = (const float*)d_in[8];
  const float* Wo = (const float*)d_in[9];
  const float* bo = (const float*)d_in[10];

  bf16* Qh = (bf16*)d_ws;
  bf16* Kh = Qh + kHeadElems;
  bf16* Vh = Kh + kHeadElems;
  bf16* Xa = Vh + kHeadElems;   // [B, S, D] bf16

  const dim3 gridP(kD / 128, kM / 64);   // (6, 64)
  const dim3 blockP(256);

  gemm_proj<0, float><<<gridP, blockP, 0, stream>>>(q, Wq, bq, Qh);
  gemm_proj<0, float><<<gridP, blockP, 0, stream>>>(k, Wk, bk, Kh);
  gemm_proj<0, float><<<gridP, blockP, 0, stream>>>(v, Wv, bv, Vh);

  attn_kernel<<<dim3(kS / 64, kB * kH), dim3(128), 0, stream>>>(Qh, Kh, Vh, Xa);

  gemm_proj<1, bf16><<<gridP, blockP, 0, stream>>>(Xa, Wo, bo, d_out);
}
